// HDRNetMini_63745904607741
// MI455X (gfx1250) — matrix operand order repaired and checked by reading, not compiled, not run on hardware
//
#include <hip/hip_runtime.h>

typedef _Float16 h16;
typedef __attribute__((ext_vector_type(16))) _Float16 v16h;
typedef __attribute__((ext_vector_type(8)))  _Float16 v8h;
typedef __attribute__((ext_vector_type(8)))  float    v8f;

__device__ __forceinline__ v8f wmma32f16(v16h a, v16h b, v8f c) {
  // D = A(16x32 f16) * B(32x16 f16) + C(16x16 f32)
  return __builtin_amdgcn_wmma_f32_16x16x32_f16(false, a, false, b, (short)0, c, false, false);
}
__device__ __forceinline__ v16h cat16(v8h a, v8h b) {
  return __builtin_shufflevector(a, b, 0,1,2,3,4,5,6,7,8,9,10,11,12,13,14,15);
}
// gfx1250 async global->LDS copy (ASYNCcnt-tracked, bypasses VGPRs).
// GV mode: operand0 = LDS byte-offset VGPR, operand1 = 64-bit global addr VGPR pair.
__device__ __forceinline__ void async_ld_b128(const void* gptr, void* lptr) {
  asm volatile("global_load_async_to_lds_b128 %0, %1, off"
               :: "v"((unsigned)(unsigned long long)lptr),
                  "v"((unsigned long long)gptr)
               : "memory");
}
__device__ __forceinline__ void wait_async0() {
  asm volatile("s_wait_asynccnt 0x0" ::: "memory");
}

// ---- weight preprocessing: f32 [OC][IC][3][3] -> f16 [tap][OC][IC] (one-shot) ----
__global__ void k_wcvt_t(const float* __restrict__ src, h16* __restrict__ dst, int OC, int IC) {
  int idx = blockIdx.x * blockDim.x + threadIdx.x;
  if (idx >= OC * IC * 9) return;
  int ic = idx % IC; int oc = (idx / IC) % OC; int tap = idx / (IC * OC);
  dst[idx] = (h16)src[(oc * IC + ic) * 9 + tap];
}
__global__ void k_wcvt(const float* __restrict__ src, h16* __restrict__ dst, int n) {
  int idx = blockIdx.x * blockDim.x + threadIdx.x;
  if (idx < n) dst[idx] = (h16)src[idx];
}

// ---------------- 1) antialiased 4x downsample 1024 -> 256 (triangle kernel) -------
__global__ void k_resize(const float* __restrict__ x, float* __restrict__ xl) {
  int idx = blockIdx.x * blockDim.x + threadIdx.x;
  if (idx >= 2 * 3 * 256 * 256) return;
  int ox = idx & 255, oy = (idx >> 8) & 255, bc = idx >> 16;
  const float w8[8] = {1.f/32, 3.f/32, 5.f/32, 7.f/32, 7.f/32, 5.f/32, 3.f/32, 1.f/32};
  const float* src = x + (size_t)bc * 1024 * 1024;
  int iy0 = oy * 4 - 2, ix0 = ox * 4 - 2;
  float acc = 0.f;
  for (int dy = 0; dy < 8; ++dy) {
    int iy = iy0 + dy; iy = iy < 0 ? 0 : (iy > 1023 ? 1023 : iy);
    float ra = 0.f;
    for (int dx = 0; dx < 8; ++dx) {
      int ix = ix0 + dx; ix = ix < 0 ? 0 : (ix > 1023 ? 1023 : ix);
      ra += w8[dx] * src[iy * 1024 + ix];
    }
    acc += w8[dy] * ra;
  }
  xl[idx] = acc;
}

// ------------- 2) low1: conv3x3 3->32 s2 + BN + ReLU, NHWC f16 out ----------------
__global__ void k_low1(const float* __restrict__ xl, const float* __restrict__ W,
                       const float* __restrict__ bconv, const float* __restrict__ gamma,
                       const float* __restrict__ beta, const float* __restrict__ mean,
                       const float* __restrict__ var, h16* __restrict__ out) {
  __shared__ float sW[32 * 27];
  for (int i = threadIdx.x; i < 32 * 27; i += blockDim.x) sW[i] = W[i];
  __syncthreads();
  int idx = blockIdx.x * blockDim.x + threadIdx.x;
  if (idx >= 2 * 128 * 128) return;
  int ox = idx & 127, oy = (idx >> 7) & 127, b = idx >> 14;
  float acc[32];
  #pragma unroll
  for (int o = 0; o < 32; ++o) acc[o] = 0.f;
  for (int ic = 0; ic < 3; ++ic)
    for (int ky = 0; ky < 3; ++ky) {
      int iy = oy * 2 - 1 + ky; if (iy < 0 || iy > 255) continue;
      for (int kx = 0; kx < 3; ++kx) {
        int ix = ox * 2 - 1 + kx; if (ix < 0 || ix > 255) continue;
        float v = xl[((size_t)(b * 3 + ic) * 256 + iy) * 256 + ix];
        #pragma unroll
        for (int o = 0; o < 32; ++o) acc[o] += v * sW[((o * 3 + ic) * 3 + ky) * 3 + kx];
      }
    }
  h16* op = out + (size_t)idx * 32;   // NHWC
  #pragma unroll
  for (int o = 0; o < 32; ++o) {
    float s = gamma[o] * rsqrtf(var[o] + 1e-5f);
    float v = (acc[o] + bconv[o] - mean[o]) * s + beta[o];
    op[o] = (h16)(v > 0.f ? v : 0.f);
  }
}

// -- 3,4,5) WMMA implicit-GEMM conv3x3 outC=64, NHWC f16 io, async-LDS staging -----
template <int INC, int STRIDE>
__global__ void k_convwmma(const h16* __restrict__ X, const h16* __restrict__ W16,
                           const float* __restrict__ bconv, const float* __restrict__ gamma,
                           const float* __restrict__ beta, const float* __restrict__ mean,
                           const float* __restrict__ var, h16* __restrict__ out,
                           int Hin, int Win, int Ho, int Wo) {
  constexpr int ICOLS = 16 * STRIDE + 2;
  __shared__ h16 tile[3 * ICOLS * INC];          // [row][col][ic]
  int blk = blockIdx.x;
  int pg = blk % (Wo / 16); int tmp = blk / (Wo / 16);
  int oy = tmp % Ho; int b = tmp / Ho;
  int px0 = pg * 16;
  int ix0 = px0 * STRIDE - 1, iy0 = oy * STRIDE - 1;
  constexpr int NU = 3 * ICOLS * (INC / 8);
  for (int u = threadIdx.x; u < NU; u += blockDim.x) {
    int c8 = u % (INC / 8); int col = (u / (INC / 8)) % ICOLS; int r = u / ((INC / 8) * ICOLS);
    int iy = iy0 + r, ix = ix0 + col;
    h16* lp = &tile[(r * ICOLS + col) * INC + c8 * 8];
    if (iy >= 0 && iy < Hin && ix >= 0 && ix < Win) {
      async_ld_b128(&X[(((size_t)b * Hin + iy) * Win + ix) * INC + c8 * 8], lp);
    } else {
      v8h z = {};
      *(v8h*)lp = z;
    }
  }
  wait_async0();
  __syncthreads();
  int wave = threadIdx.x >> 5;                   // output-channel tile (0..3)
  int lane = threadIdx.x & 31;
  int n = lane & 15;                             // pixel col / A row (M)
  int abase = (lane < 16) ? 0 : 8;               // ISA 16-bit A 16x32 K-base
  int bbase = (lane < 16) ? 0 : 8;               // ISA 16-bit B 32x16 K-base
  v8f acc = {};
  constexpr int NCHUNK = 9 * INC / 32;
  #pragma unroll
  for (int c = 0; c < NCHUNK; ++c) {
    int tap, icb;
    if (INC == 32) { tap = c; icb = 0; } else { tap = c >> 1; icb = (c & 1) * 32; }
    int ty = tap / 3, tx = tap % 3;
    const h16* ap = W16 + ((size_t)tap * 64 + wave * 16 + n) * INC + icb + abase;
    v16h af = cat16(*(const v8h*)ap, *(const v8h*)(ap + 16));
    const h16* bp = &tile[(ty * ICOLS + n * STRIDE + tx) * INC + icb + bbase];
    v16h bf = cat16(*(const v8h*)bp, *(const v8h*)(bp + 16));
    acc = wmma32f16(af, bf, acc);
  }
  int px = px0 + n;
  v8h ov;
  #pragma unroll
  for (int r = 0; r < 8; ++r) {
    int oc = wave * 16 + abase + r;              // C layout: lane<16 -> M=r else M=r+8
    float s = gamma[oc] * rsqrtf(var[oc] + 1e-5f);
    float v = (acc[r] + bconv[oc] - mean[oc]) * s + beta[oc];
    ov[r] = (h16)(v > 0.f ? v : 0.f);
  }
  *(v8h*)&out[(((size_t)b * Ho + oy) * Wo + px) * 64 + wave * 16 + abase] = ov;
}

// -------- 6) head: 1x1 conv 64->96 WMMA GEMM; grid out fp32 [b][pix][96] ----------
__global__ void k_head(const h16* __restrict__ X, const h16* __restrict__ W16,
                       const float* __restrict__ bias, float* __restrict__ grid) {
  int blk = blockIdx.x;                          // B * 6 * 256 waves
  int pt = blk % 256; int tmp = blk / 256;
  int mt = tmp % 6; int b = tmp / 6;
  int lane = threadIdx.x & 31;
  int n = lane & 15;
  int abase = (lane < 16) ? 0 : 8, bbase = (lane < 16) ? 0 : 8;
  v8f acc = {};
  #pragma unroll
  for (int c = 0; c < 2; ++c) {
    const h16* ap = W16 + (size_t)(mt * 16 + n) * 64 + c * 32 + abase;
    v16h af = cat16(*(const v8h*)ap, *(const v8h*)(ap + 16));
    const h16* bp = X + ((size_t)b * 4096 + pt * 16 + n) * 64 + c * 32 + bbase;
    v16h bf = cat16(*(const v8h*)bp, *(const v8h*)(bp + 16));
    acc = wmma32f16(af, bf, acc);
  }
  v8f vout;
  #pragma unroll
  for (int r = 0; r < 8; ++r) vout[r] = acc[r] + bias[mt * 16 + abase + r];
  *(v8f*)&grid[((size_t)b * 4096 + pt * 16 + n) * 96 + mt * 16 + abase] = vout;
}

// --------------- 7) g1: conv3x3 3->16 + BN + ReLU, NHWC f16 out -------------------
__global__ void k_g1(const float* __restrict__ x, const float* __restrict__ W,
                     const float* __restrict__ bconv, const float* __restrict__ gamma,
                     const float* __restrict__ beta, const float* __restrict__ mean,
                     const float* __restrict__ var, h16* __restrict__ out) {
  __shared__ float sW[16 * 27];
  __shared__ float sS[16], sB[16];
  for (int i = threadIdx.x; i < 16 * 27; i += blockDim.x) sW[i] = W[i];
  if (threadIdx.x < 16) {
    int o = threadIdx.x;
    float s = gamma[o] * rsqrtf(var[o] + 1e-5f);
    sS[o] = s; sB[o] = (bconv[o] - mean[o]) * s + beta[o];
  }
  __syncthreads();
  int idx = blockIdx.x * blockDim.x + threadIdx.x;
  if (idx >= 2 * 1024 * 1024) return;
  int px = idx & 1023, py = (idx >> 10) & 1023, b = idx >> 20;
  float acc[16];
  #pragma unroll
  for (int o = 0; o < 16; ++o) acc[o] = 0.f;
  for (int ic = 0; ic < 3; ++ic)
    for (int ky = 0; ky < 3; ++ky) {
      int iy = py - 1 + ky; if (iy < 0 || iy > 1023) continue;
      for (int kx = 0; kx < 3; ++kx) {
        int ix = px - 1 + kx; if (ix < 0 || ix > 1023) continue;
        float v = x[((size_t)(b * 3 + ic) * 1024 + iy) * 1024 + ix];
        #pragma unroll
        for (int o = 0; o < 16; ++o) acc[o] += v * sW[((o * 3 + ic) * 3 + ky) * 3 + kx];
      }
    }
  h16* op = out + (size_t)idx * 16;              // NHWC
  #pragma unroll
  for (int o = 0; o < 16; ++o) {
    float v = acc[o] * sS[o] + sB[o];
    op[o] = (h16)(v > 0.f ? v : 0.f);
  }
}

// ---- 8) fused: g2 (WMMA conv 16->16) + g3 sigmoid + trilinear slice + apply ------
__global__ void k_g2_fused(const h16* __restrict__ G1, const h16* __restrict__ W16,
                           const float* __restrict__ b2, const float* __restrict__ gamma,
                           const float* __restrict__ beta, const float* __restrict__ mean,
                           const float* __restrict__ var,
                           const float* __restrict__ w3, const float* __restrict__ b3,
                           const float* __restrict__ grid, const float* __restrict__ x,
                           float* __restrict__ out) {
  __shared__ h16 tile[3 * 66 * 16];              // [row][col][ic]
  int blk = blockIdx.x;                          // B * 1024 * 16 blocks (64 px each)
  int cg = blk & 15; int tmp = blk >> 4;
  int y = tmp & 1023; int b = tmp >> 10;
  int col0 = cg * 64 - 1;
  for (int u = threadIdx.x; u < 3 * 66 * 2; u += blockDim.x) {
    int h8 = u & 1; int col = (u >> 1) % 66; int r = u / (66 * 2);
    int iy = y - 1 + r, ix = col0 + col;
    h16* lp = &tile[(r * 66 + col) * 16 + h8 * 8];
    if (iy >= 0 && iy < 1024 && ix >= 0 && ix < 1024) {
      async_ld_b128(&G1[(((size_t)b * 1024 + iy) * 1024 + ix) * 16 + h8 * 8], lp);
    } else {
      v8h z = {};
      *(v8h*)lp = z;
    }
  }
  wait_async0();
  int wave = threadIdx.x >> 5, lane = threadIdx.x & 31;
  int n = lane & 15;
  // prefetch the RGB pixels consumed after the WMMA chunks
  {
    size_t pix = (size_t)y * 1024 + (cg * 64 + wave * 16 + n);
    __builtin_prefetch(&x[(size_t)(b * 3) * 1048576 + pix], 0, 3);
  }
  __syncthreads();
  int abase = (lane < 16) ? 0 : 8, bbase = (lane < 16) ? 0 : 16;
  v8h z8 = {};
  v8f acc = {};
  #pragma unroll
  for (int c = 0; c < 5; ++c) {                  // K = 9 taps * 16ch, padded to 160
    int tapA = 2 * c + (abase >> 3);
    const h16* ap0 = W16 + ((size_t)tapA * 16 + n) * 16;
    v8h a0 = (tapA < 9) ? *(const v8h*)ap0 : z8;
    v8h a1 = (tapA < 9) ? *(const v8h*)(ap0 + 8) : z8;
    v16h af = cat16(a0, a1);
    int tapB = 2 * c + (bbase >> 4);
    v16h bf;
    if (tapB < 9) {
      int ty = tapB / 3, tx = tapB % 3;
      const h16* bp = &tile[(ty * 66 + wave * 16 + n + tx) * 16];
      bf = cat16(*(const v8h*)bp, *(const v8h*)(bp + 8));
    } else {
      bf = cat16(z8, z8);
    }
    acc = wmma32f16(af, bf, acc);
  }
  // BN + ReLU + g3 dot (channels split across lane L and L^16 -> shfl reduce)
  float part = 0.f;
  #pragma unroll
  for (int r = 0; r < 8; ++r) {
    int oc = abase + r;
    float s = gamma[oc] * rsqrtf(var[oc] + 1e-5f);
    float v = (acc[r] + b2[oc] - mean[oc]) * s + beta[oc];
    v = v > 0.f ? v : 0.f;
    part += v * w3[oc];
  }
  part += __shfl_xor(part, 16, 32);
  if (lane < 16) {
    float g = 1.f / (1.f + __expf(-(part + b3[0])));
    int px = cg * 64 + wave * 16 + n;
    float ys = (float)y * (63.f / (1023.f + 1e-8f));
    float xs = (float)px * (63.f / (1023.f + 1e-8f));
    float zs = g * 7.f;
    float y0f = floorf(ys); y0f = y0f < 0.f ? 0.f : (y0f > 63.f ? 63.f : y0f);
    float x0f = floorf(xs); x0f = x0f < 0.f ? 0.f : (x0f > 63.f ? 63.f : x0f);
    float z0f = floorf(zs); z0f = z0f < 0.f ? 0.f : (z0f > 7.f ? 7.f : z0f);
    int y0 = (int)y0f, x0 = (int)x0f, z0 = (int)z0f;
    int y1 = y0 + 1 > 63 ? 63 : y0 + 1;
    int x1 = x0 + 1 > 63 ? 63 : x0 + 1;
    int z1 = z0 + 1 > 7 ? 7 : z0 + 1;
    float wy = ys - y0f, wx = xs - x0f, wz = zs - z0f;
    int cy[8] = {y0, y0, y0, y0, y1, y1, y1, y1};
    int cx[8] = {x0, x0, x1, x1, x0, x0, x1, x1};
    int cz[8] = {z0, z1, z0, z1, z0, z1, z0, z1};
    float cw[8] = {(1 - wy) * (1 - wx) * (1 - wz), (1 - wy) * (1 - wx) * wz,
                   (1 - wy) * wx * (1 - wz),       (1 - wy) * wx * wz,
                   wy * (1 - wx) * (1 - wz),       wy * (1 - wx) * wz,
                   wy * wx * (1 - wz),             wy * wx * wz};
    float coeff[12];
    #pragma unroll
    for (int i = 0; i < 12; ++i) coeff[i] = 0.f;
    #pragma unroll
    for (int k = 0; k < 8; ++k) {
      const float* gp = grid + ((size_t)b * 4096 + cy[k] * 64 + cx[k]) * 96;
      float wgt = cw[k];
      #pragma unroll
      for (int i = 0; i < 12; ++i) coeff[i] += wgt * gp[i * 8 + cz[k]];
    }
    size_t pix = (size_t)y * 1024 + px;
    float rr = x[(size_t)(b * 3 + 0) * 1048576 + pix];
    float gg = x[(size_t)(b * 3 + 1) * 1048576 + pix];
    float bb = x[(size_t)(b * 3 + 2) * 1048576 + pix];
    #pragma unroll
    for (int c = 0; c < 3; ++c) {
      float v = coeff[c * 4] * rr + coeff[c * 4 + 1] * gg + coeff[c * 4 + 2] * bb + coeff[c * 4 + 3];
      v = v < 0.f ? 0.f : (v > 1.f ? 1.f : v);
      out[(size_t)(b * 3 + c) * 1048576 + pix] = v;
    }
  }
}

extern "C" void kernel_launch(void* const* d_in, const int* in_sizes, int n_in,
                              void* d_out, int out_size, void* d_ws, size_t ws_size,
                              hipStream_t stream) {
  (void)n_in; (void)out_size; (void)ws_size;
  auto f = [&](int i) { return (const float*)d_in[i]; };
  const float *x, *l1w, *l1b, *l1g, *l1be, *l1m, *l1v;
  const float *l2w, *l2b, *l2g, *l2be, *l2m, *l2v;
  const float *l3w, *l3b, *l3g, *l3be, *l3m, *l3v;
  const float *l4w, *l4b, *l4g, *l4be, *l4m, *l4v;
  const float *hw, *hb, *g1w, *g1b, *g1g, *g1be, *g1m, *g1v;
  const float *g2w, *g2b, *g2g, *g2be, *g2m, *g2v, *g3w, *g3b;
  if (in_sizes[0] == 2 * 3 * 1024 * 1024) {
    // insertion order: x, low1..low4, head, g1..g3 (each: w,b,[gamma,beta,mean,var])
    x = f(0);
    l1w = f(1);  l1b = f(2);  l1g = f(3);  l1be = f(4);  l1m = f(5);  l1v = f(6);
    l2w = f(7);  l2b = f(8);  l2g = f(9);  l2be = f(10); l2m = f(11); l2v = f(12);
    l3w = f(13); l3b = f(14); l3g = f(15); l3be = f(16); l3m = f(17); l3v = f(18);
    l4w = f(19); l4b = f(20); l4g = f(21); l4be = f(22); l4m = f(23); l4v = f(24);
    hw = f(25);  hb = f(26);
    g1w = f(27); g1b = f(28); g1g = f(29); g1be = f(30); g1m = f(31); g1v = f(32);
    g2w = f(33); g2b = f(34); g2g = f(35); g2be = f(36); g2m = f(37); g2v = f(38);
    g3w = f(39); g3b = f(40);
  } else {
    // jax pytree sorted-key order: params{g1,g2,g3,head,low1..4}, then x
    g1b = f(0);  g1be = f(1);  g1g = f(2);  g1m = f(3);  g1v = f(4);  g1w = f(5);
    g2b = f(6);  g2be = f(7);  g2g = f(8);  g2m = f(9);  g2v = f(10); g2w = f(11);
    g3b = f(12); g3w = f(13);
    hb = f(14);  hw = f(15);
    l1b = f(16); l1be = f(17); l1g = f(18); l1m = f(19); l1v = f(20); l1w = f(21);
    l2b = f(22); l2be = f(23); l2g = f(24); l2m = f(25); l2v = f(26); l2w = f(27);
    l3b = f(28); l3be = f(29); l3g = f(30); l3m = f(31); l3v = f(32); l3w = f(33);
    l4b = f(34); l4be = f(35); l4g = f(36); l4m = f(37); l4v = f(38); l4w = f(39);
    x = f(40);
  }

  char* ws = (char*)d_ws;
  float* xl = (float*)ws;    ws += (size_t)2 * 3 * 256 * 256 * 4;     // 1.5 MB
  h16* l1o = (h16*)ws;       ws += (size_t)2 * 128 * 128 * 32 * 2;    // 2 MB   NHWC
  h16* l2o = (h16*)ws;       ws += (size_t)2 * 64 * 64 * 64 * 2;      // 1 MB   NHWC
  h16* l3o = (h16*)ws;       ws += (size_t)2 * 64 * 64 * 64 * 2;      // 1 MB   NHWC
  h16* l4o = (h16*)ws;       ws += (size_t)2 * 64 * 64 * 64 * 2;      // 1 MB   NHWC
  float* gridp = (float*)ws; ws += (size_t)2 * 4096 * 96 * 4;         // 3 MB   [b][pix][96]
  h16* g1o = (h16*)ws;       ws += (size_t)2 * 1024 * 1024 * 16 * 2;  // 67 MB  NHWC
  h16* w2h = (h16*)ws;       ws += (size_t)9 * 64 * 32 * 2;           // [tap][oc][ic]
  h16* w3h = (h16*)ws;       ws += (size_t)9 * 64 * 64 * 2;
  h16* w4h = (h16*)ws;       ws += (size_t)9 * 64 * 64 * 2;
  h16* whh = (h16*)ws;       ws += (size_t)96 * 64 * 2;               // [oc][ic]
  h16* wg2h = (h16*)ws;      ws += (size_t)9 * 16 * 16 * 2;           // [tap][oc][ic]
  float* out = (float*)d_out;

  k_wcvt_t<<<(9 * 64 * 32 + 255) / 256, 256, 0, stream>>>(l2w, w2h, 64, 32);
  k_wcvt_t<<<(9 * 64 * 64 + 255) / 256, 256, 0, stream>>>(l3w, w3h, 64, 64);
  k_wcvt_t<<<(9 * 64 * 64 + 255) / 256, 256, 0, stream>>>(l4w, w4h, 64, 64);
  k_wcvt_t<<<(9 * 16 * 16 + 255) / 256, 256, 0, stream>>>(g2w, wg2h, 16, 16);
  k_wcvt<<<(96 * 64 + 255) / 256, 256, 0, stream>>>(hw, whh, 96 * 64);

  k_resize<<<(2 * 3 * 256 * 256 + 255) / 256, 256, 0, stream>>>(x, xl);
  k_low1<<<(2 * 128 * 128 + 255) / 256, 256, 0, stream>>>(xl, l1w, l1b, l1g, l1be, l1m, l1v, l1o);
  k_convwmma<32, 2><<<2 * 64 * 4, 128, 0, stream>>>(l1o, w2h, l2b, l2g, l2be, l2m, l2v, l2o,
                                                    128, 128, 64, 64);
  k_convwmma<64, 1><<<2 * 64 * 4, 128, 0, stream>>>(l2o, w3h, l3b, l3g, l3be, l3m, l3v, l3o,
                                                    64, 64, 64, 64);
  k_convwmma<64, 1><<<2 * 64 * 4, 128, 0, stream>>>(l3o, w4h, l4b, l4g, l4be, l4m, l4v, l4o,
                                                    64, 64, 64, 64);
  k_head<<<2 * 6 * 256, 32, 0, stream>>>(l4o, whh, hb, gridp);
  k_g1<<<(2 * 1024 * 1024 + 255) / 256, 256, 0, stream>>>(x, g1w, g1b, g1g, g1be, g1m, g1v, g1o);
  k_g2_fused<<<2 * 1024 * 16, 128, 0, stream>>>(g1o, wg2h, g2b, g2g, g2be, g2m, g2v,
                                                g3w, g3b, gridp, x, out);
}
